// GNN_MLP_VariationalAutoEncoder_29016799052368
// MI455X (gfx1250) — matrix operand order repaired and checked by reading, not compiled, not run on hardware
//
#include <hip/hip_runtime.h>
#include <hip/hip_bf16.h>

// ---------------- problem constants (from reference) ----------------
#define N_NODES 20000
#define N_EDGES 320000
#define H_DIM   512
#define OUT_DIM 1024
#define POOL_P  100      // partial-reduction blocks for pooling
#define POOL_ROWS 200    // N_NODES / POOL_P

// ---------------- WMMA types (CDNA5 / gfx1250, wave32) ----------------
typedef __attribute__((ext_vector_type(16))) __bf16 v16bf;
typedef __attribute__((ext_vector_type(8)))  float  v8f;

union BFrag { v16bf v; unsigned long long q[4]; };

__device__ __forceinline__ __bf16 f2bf(float f) {
  // round-to-nearest-even f32 -> bf16
  union { float f; unsigned u; } in; in.f = f;
  unsigned u = in.u;
  unsigned r = u + 0x7FFFu + ((u >> 16) & 1u);
  union { unsigned short s; __bf16 b; } out;
  out.s = (unsigned short)(r >> 16);
  return out.b;
}

// =====================================================================
// GEMM: C[M,Nd] = A[M,K] @ B[K,Nd] (+bias).
//  - A: bf16 [M,K] row-major (activations, pre-converted)
//  - Bt: bf16 [Nd,K] (weights pre-transposed+converted) so every LDS tile
//    chunk is a contiguous 16B copy -> async global->LDS B128 path.
//  - f32 accumulate with v_wmma_f32_16x16x32_bf16.
// Block tile 32(M) x 128(N), K-step 32, 256 threads = 8 waves,
// each wave computes one 16x32 sub-tile (2 WMMA accumulators).
// Requires: M % 32 == 0, Nd % 128 == 0, K % 32 == 0 (true for all calls).
// =====================================================================
__global__ void __launch_bounds__(256)
gemm_bf16_wmma_async(const __bf16* __restrict__ A, const __bf16* __restrict__ Bt,
                     float* __restrict__ C, const float* __restrict__ bias,
                     int M, int Nd, int K) {
  __shared__ __attribute__((aligned(16))) __bf16 sA[32 * 32];    // [m][k], row = 64B
  __shared__ __attribute__((aligned(16))) __bf16 sB[128 * 32];   // [n][k], row = 64B

  const int bm   = blockIdx.y;
  const int bn   = blockIdx.x;
  const int tid  = threadIdx.x;
  const int wave = tid >> 5;
  const int lane = tid & 31;
  const int wm   = (wave >> 2) << 4;   // 0 or 16
  const int wn   = (wave & 3) << 5;    // 0,32,64,96
  const int l15  = lane & 15;          // m (A frag), n (B frag), col (C frag)
  const int half = lane >> 4;          // K-half selector per ISA layout

  // LDS byte addresses (low 32 bits of generic shared pointer = LDS offset)
  const unsigned sAb = (unsigned)(size_t)(void*)sA;
  const unsigned sBb = (unsigned)(size_t)(void*)sB;

  // async staging: 16B chunks. sA = 32 rows x 4 chunks = 128 (tid<128, 1 each);
  // sB = 128 rows x 4 chunks = 512 (2 per thread).
  const int rA = tid >> 2, cA = tid & 3;
  const unsigned ldsA  = sAb + rA * 64 + cA * 16;
  const int nB0 = tid >> 2,          sB0c = tid & 3;
  const int nB1 = (tid + 256) >> 2,  sB1c = (tid + 256) & 3;
  const unsigned ldsB0 = sBb + nB0 * 64 + sB0c * 16;
  const unsigned ldsB1 = sBb + nB1 * 64 + sB1c * 16;

  const __bf16* gA  = A  + (size_t)(bm * 32 + rA) * K + cA * 8;
  const __bf16* gB0 = Bt + (size_t)(bn * 128 + nB0) * K + sB0c * 8;
  const __bf16* gB1 = Bt + (size_t)(bn * 128 + nB1) * K + sB1c * 8;

  v8f acc0 = {}; v8f acc1 = {};

  for (int k0 = 0; k0 < K; k0 += 32) {
    __syncthreads();   // previous compute done before overwriting LDS
    // ---- async global -> LDS staging (ASYNCcnt), no VGPR round-trip ----
    if (tid < 128) {
      asm volatile("global_load_async_to_lds_b128 %0, %1, off"
                   :: "v"(ldsA), "v"((unsigned long long)(gA + k0)) : "memory");
    }
    asm volatile("global_load_async_to_lds_b128 %0, %1, off"
                 :: "v"(ldsB0), "v"((unsigned long long)(gB0 + k0)) : "memory");
    asm volatile("global_load_async_to_lds_b128 %0, %1, off"
                 :: "v"(ldsB1), "v"((unsigned long long)(gB1 + k0)) : "memory");
    // prefetch next K-tile into GL2 while async copies land
    if (k0 + 32 < K) {
      __builtin_prefetch(gB0 + k0 + 32, 0, 0);
      if (tid < 128) __builtin_prefetch(gA + k0 + 32, 0, 0);
    }
    asm volatile("s_wait_asynccnt 0x0" ::: "memory");
    __syncthreads();   // all waves' tiles visible

    // A fragment: lane(half, m=l15): elems j=0..7 -> k = half*8 + j,
    //             elems j=8..15 -> k = 16 + half*8 + (j-8)   (ISA 16-bit A 16x32)
    BFrag a;
    {
      const unsigned long long* p0 =
          (const unsigned long long*)&sA[(wm + l15) * 32 + half * 8];
      const unsigned long long* p1 =
          (const unsigned long long*)&sA[(wm + l15) * 32 + half * 8 + 16];
      a.q[0] = p0[0]; a.q[1] = p0[1]; a.q[2] = p1[0]; a.q[3] = p1[1];
    }
    // B fragments: lane(half, n=l15): elems j=0..7 -> k = half*8 + j, j=8..15 -> k = 16 + half*8 + (j-8)
    BFrag b0, b1;
    {
      const unsigned long long* p =
          (const unsigned long long*)&sB[(wn + l15) * 32 + half * 8];
      b0.q[0] = p[0]; b0.q[1] = p[1]; b0.q[2] = p[4]; b0.q[3] = p[5];
      const unsigned long long* p2 =
          (const unsigned long long*)&sB[(wn + 16 + l15) * 32 + half * 8];
      b1.q[0] = p2[0]; b1.q[1] = p2[1]; b1.q[2] = p2[4]; b1.q[3] = p2[5];
    }

    acc0 = __builtin_amdgcn_wmma_f32_16x16x32_bf16(
        false, a.v, false, b0.v, (short)0, acc0, false, false);
    acc1 = __builtin_amdgcn_wmma_f32_16x16x32_bf16(
        false, a.v, false, b1.v, (short)0, acc1, false, false);
  }

  // C/D layout: lane -> col = l15, rows m = (lane>>4)*8 + r
  const int mbase = half << 3;
  for (int r = 0; r < 8; ++r) {
    int row = bm * 32 + wm + mbase + r;
    int c0  = bn * 128 + wn + l15;
    int c1  = c0 + 16;
    float v0 = acc0[r], v1 = acc1[r];
    if (bias) { v0 += bias[c0]; v1 += bias[c1]; }
    C[(size_t)row * Nd + c0] = v0;
    C[(size_t)row * Nd + c1] = v1;
  }
}

// ---------------- conversion kernels ----------------
__global__ void k_f2bf(const float* __restrict__ in, __bf16* __restrict__ out, int n) {
  int i = blockIdx.x * blockDim.x + threadIdx.x;
  if (i < n) out[i] = f2bf(in[i]);
}

// W [K,Nd] f32 -> Wt [Nd,K] bf16 (transpose + convert, once per launch)
__global__ void k_wtrans(const float* __restrict__ W, __bf16* __restrict__ Wt,
                         int K, int Nd) {
  int i = blockIdx.x * blockDim.x + threadIdx.x;
  if (i < K * Nd) {
    int k = i / Nd, n = i - k * Nd;
    Wt[(size_t)n * K + k] = f2bf(W[i]);
  }
}

// ---------------- graph preprocessing ----------------
__global__ void k_zero(float* __restrict__ p, int n) {
  int i = blockIdx.x * blockDim.x + threadIdx.x;
  if (i < n) p[i] = 0.0f;
}

__global__ void k_deg_acc(float* __restrict__ deg, const int* __restrict__ dst,
                          const float* __restrict__ ew, int e) {
  int i = blockIdx.x * blockDim.x + threadIdx.x;
  if (i < e) atomicAdd(&deg[dst[i]], ew[i]);
}

__global__ void k_dinv(float* __restrict__ deg, int n) {
  int i = blockIdx.x * blockDim.x + threadIdx.x;
  if (i < n) deg[i] = rsqrtf(deg[i] + 1.0f);   // in-place deg -> dinv
}

// agg[n][c] = dinv[n]^2 * xw[n][c] + b[c]   (self-loop term + bias)
__global__ void k_self_init(float* __restrict__ agg, const float* __restrict__ xw,
                            const float* __restrict__ dinv, const float* __restrict__ b,
                            int od, int total) {
  int i = blockIdx.x * blockDim.x + threadIdx.x;
  if (i < total) {
    int n = i / od, c = i - n * od;
    float di = dinv[n];
    agg[i] = di * di * xw[i] + b[c];
  }
}

// agg[dst] += dinv[src]*ew*dinv[dst] * xw[src]
__global__ void k_edge_scatter(float* __restrict__ agg, const float* __restrict__ xw,
                               const int* __restrict__ src, const int* __restrict__ dst,
                               const float* __restrict__ ew, const float* __restrict__ dinv,
                               int od) {
  int e = blockIdx.x;
  int c = blockIdx.y * blockDim.x + threadIdx.x;
  int s = src[e], d = dst[e];
  float nrm = dinv[s] * ew[e] * dinv[d];
  atomicAdd(&agg[(size_t)d * od + c], nrm * xw[(size_t)s * od + c]);
}

// row-wise: out_bf16 = relu(h / max(||h||2, 1e-12)); one block per row.
// Output is bf16: it is only ever consumed as the next GEMM's A operand.
__global__ void __launch_bounds__(256)
k_l2norm_relu_bf(const float* __restrict__ h, __bf16* __restrict__ out, int od) {
  __shared__ float red[256];
  size_t base = (size_t)blockIdx.x * od;
  float ss = 0.0f;
  for (int c = threadIdx.x; c < od; c += 256) { float v = h[base + c]; ss += v * v; }
  red[threadIdx.x] = ss;
  __syncthreads();
  for (int s = 128; s > 0; s >>= 1) {
    if (threadIdx.x < s) red[threadIdx.x] += red[threadIdx.x + s];
    __syncthreads();
  }
  float scale = 1.0f / fmaxf(sqrtf(red[0]), 1e-12f);
  for (int c = threadIdx.x; c < od; c += 256) {
    float v = h[base + c] * scale;
    out[base + c] = f2bf(v > 0.0f ? v : 0.0f);
  }
}

// fused reparameterize + partial max/sum pooling over a row chunk
__global__ void __launch_bounds__(512)
k_pool_partial(const float* __restrict__ mu, const float* __restrict__ lv,
               const float* __restrict__ eps, const float* __restrict__ beta,
               float* __restrict__ pmax, float* __restrict__ psum) {
  int c = threadIdx.x;                 // 0..511 (H_DIM)
  int r0 = blockIdx.x * POOL_ROWS;
  float b = beta[0];
  float mx = -3.402823466e38f, sm = 0.0f;
  for (int r = 0; r < POOL_ROWS; ++r) {
    size_t idx = (size_t)(r0 + r) * H_DIM + c;
    float z = mu[idx] + 0.01f * eps[idx] * expf(0.5f * b * lv[idx]);
    mx = fmaxf(mx, z);
    sm += z;
  }
  pmax[blockIdx.x * H_DIM + c] = mx;
  psum[blockIdx.x * H_DIM + c] = sm;
}

// final reduce -> reduce_z = [max(512) | mean(512) | y_target]
__global__ void __launch_bounds__(512)
k_pool_final(const float* __restrict__ pmax, const float* __restrict__ psum,
             const float* __restrict__ y_t, float* __restrict__ rz) {
  int c = threadIdx.x;
  float mx = -3.402823466e38f, sm = 0.0f;
  for (int p = 0; p < POOL_P; ++p) {
    mx = fmaxf(mx, pmax[p * H_DIM + c]);
    sm += psum[p * H_DIM + c];
  }
  rz[c] = mx;
  rz[H_DIM + c] = sm / (float)N_NODES;
  if (c == 0) rz[2 * H_DIM] = y_t[0];
}

// 1xK @ KxNd GEMV with activation. mode 0 = relu, 1 = sigmoid
__global__ void k_decoder(const float* __restrict__ in, const float* __restrict__ W,
                          const float* __restrict__ b, float* __restrict__ out,
                          int K, int Nd, int mode) {
  int j = blockIdx.x * blockDim.x + threadIdx.x;
  if (j >= Nd) return;
  float acc = b[j];
  for (int i = 0; i < K; ++i) acc += in[i] * W[(size_t)i * Nd + j];
  if (mode == 0) out[j] = acc > 0.0f ? acc : 0.0f;
  else           out[j] = 1.0f / (1.0f + expf(-acc));
}

// =====================================================================
extern "C" void kernel_launch(void* const* d_in, const int* in_sizes, int n_in,
                              void* d_out, int out_size, void* d_ws, size_t ws_size,
                              hipStream_t stream) {
  (void)in_sizes; (void)n_in; (void)out_size; (void)ws_size;

  const float* x    = (const float*)d_in[0];
  const int*   ei   = (const int*)d_in[1];
  const float* ew   = (const float*)d_in[2];
  const float* beta = (const float*)d_in[3];
  const float* y_t  = (const float*)d_in[4];
  const float* eps  = (const float*)d_in[5];
  const float* W1 = (const float*)d_in[6];  const float* b1 = (const float*)d_in[7];
  const float* W2 = (const float*)d_in[8];  const float* b2 = (const float*)d_in[9];
  const float* W3 = (const float*)d_in[10]; const float* b3 = (const float*)d_in[11];
  const float* Wmu = (const float*)d_in[12]; const float* bmu = (const float*)d_in[13];
  const float* Wlv = (const float*)d_in[14]; const float* blv = (const float*)d_in[15];
  const float* Wd1 = (const float*)d_in[16]; const float* bd1 = (const float*)d_in[17];
  const float* Wd2 = (const float*)d_in[18]; const float* bd2 = (const float*)d_in[19];

  const int* src = ei;
  const int* dst = ei + N_EDGES;

  float* out = (float*)d_out;
  float* mu  = out + OUT_DIM;                       // [N, H]
  float* lv  = mu + (size_t)N_NODES * H_DIM;        // [N, H]

  // ---- workspace carve-out (256B-aligned regions) ----
  char* p = (char*)d_ws;
  auto alloc = [&](size_t bytes) {
    char* r = p;
    p += (bytes + 255) & ~(size_t)255;
    return r;
  };
  float*  bufA = (float*)alloc((size_t)N_NODES * 1024 * 4);   // xw (GEMM out)
  float*  bufB = (float*)alloc((size_t)N_NODES * 1024 * 4);   // agg
  __bf16* Abf  = (__bf16*)alloc((size_t)N_NODES * 1024 * 2);  // bf16 activations
  __bf16* W1t  = (__bf16*)alloc((size_t)512 * 1024 * 2);      // [1024][512]
  __bf16* W2t  = (__bf16*)alloc((size_t)1024 * 512 * 2);      // [512][1024]
  __bf16* W3t  = (__bf16*)alloc((size_t)512 * 256 * 2);       // [256][512]
  __bf16* Wmut = (__bf16*)alloc((size_t)256 * 512 * 2);       // [512][256]
  __bf16* Wlvt = (__bf16*)alloc((size_t)256 * 512 * 2);       // [512][256]
  float*  dinv = (float*)alloc((size_t)N_NODES * 4);
  float*  pmax = (float*)alloc((size_t)POOL_P * H_DIM * 4);
  float*  psum = (float*)alloc((size_t)POOL_P * H_DIM * 4);
  float*  rz   = (float*)alloc((2 * H_DIM + 1) * 4);
  float*  hdec = (float*)alloc((2 * H_DIM + 1) * 4);

  // ---- degree / symmetric normalization ----
  k_zero<<<(N_NODES + 255) / 256, 256, 0, stream>>>(dinv, N_NODES);
  k_deg_acc<<<(N_EDGES + 255) / 256, 256, 0, stream>>>(dinv, dst, ew, N_EDGES);
  k_dinv<<<(N_NODES + 255) / 256, 256, 0, stream>>>(dinv, N_NODES);

  // ---- one-time conversions: x -> bf16, weights -> bf16 transposed ----
  k_f2bf<<<(N_NODES * 512 + 255) / 256, 256, 0, stream>>>(x, Abf, N_NODES * 512);
  k_wtrans<<<(512 * 1024 + 255) / 256, 256, 0, stream>>>(W1, W1t, 512, 1024);
  k_wtrans<<<(1024 * 512 + 255) / 256, 256, 0, stream>>>(W2, W2t, 1024, 512);
  k_wtrans<<<(512 * 256 + 255) / 256, 256, 0, stream>>>(W3, W3t, 512, 256);
  k_wtrans<<<(256 * 512 + 255) / 256, 256, 0, stream>>>(Wmu, Wmut, 256, 512);
  k_wtrans<<<(256 * 512 + 255) / 256, 256, 0, stream>>>(Wlv, Wlvt, 256, 512);

  // ---- GCN layer 1: 512 -> 1024 ----
  gemm_bf16_wmma_async<<<dim3(1024 / 128, N_NODES / 32), 256, 0, stream>>>(
      Abf, W1t, bufA, nullptr, N_NODES, 1024, 512);
  k_self_init<<<(N_NODES * 1024 + 255) / 256, 256, 0, stream>>>(
      bufB, bufA, dinv, b1, 1024, N_NODES * 1024);
  k_edge_scatter<<<dim3(N_EDGES, 1024 / 256), 256, 0, stream>>>(
      bufB, bufA, src, dst, ew, dinv, 1024);
  k_l2norm_relu_bf<<<N_NODES, 256, 0, stream>>>(bufB, Abf, 1024);

  // ---- GCN layer 2: 1024 -> 512 ----
  gemm_bf16_wmma_async<<<dim3(512 / 128, N_NODES / 32), 256, 0, stream>>>(
      Abf, W2t, bufA, nullptr, N_NODES, 512, 1024);
  k_self_init<<<(N_NODES * 512 + 255) / 256, 256, 0, stream>>>(
      bufB, bufA, dinv, b2, 512, N_NODES * 512);
  k_edge_scatter<<<dim3(N_EDGES, 512 / 256), 256, 0, stream>>>(
      bufB, bufA, src, dst, ew, dinv, 512);
  k_l2norm_relu_bf<<<N_NODES, 256, 0, stream>>>(bufB, Abf, 512);

  // ---- GCN layer 3: 512 -> 256 ----
  gemm_bf16_wmma_async<<<dim3(256 / 128, N_NODES / 32), 256, 0, stream>>>(
      Abf, W3t, bufA, nullptr, N_NODES, 256, 512);
  k_self_init<<<(N_NODES * 256 + 255) / 256, 256, 0, stream>>>(
      bufB, bufA, dinv, b3, 256, N_NODES * 256);
  k_edge_scatter<<<dim3(N_EDGES, 256 / 256), 256, 0, stream>>>(
      bufB, bufA, src, dst, ew, dinv, 256);
  k_l2norm_relu_bf<<<N_NODES, 256, 0, stream>>>(bufB, Abf, 256);

  // ---- VAE heads: mu / logvar (written straight into d_out, bias fused) ----
  gemm_bf16_wmma_async<<<dim3(512 / 128, N_NODES / 32), 256, 0, stream>>>(
      Abf, Wmut, mu, bmu, N_NODES, 512, 256);
  gemm_bf16_wmma_async<<<dim3(512 / 128, N_NODES / 32), 256, 0, stream>>>(
      Abf, Wlvt, lv, blv, N_NODES, 512, 256);

  // ---- reparameterize + global max/mean pool ----
  k_pool_partial<<<POOL_P, H_DIM, 0, stream>>>(mu, lv, eps, beta, pmax, psum);
  k_pool_final<<<1, H_DIM, 0, stream>>>(pmax, psum, y_t, rz);

  // ---- decoder MLP: (2H+1)->(2H+1) relu, (2H+1)->OUT sigmoid ----
  k_decoder<<<(2 * H_DIM + 1 + 255) / 256, 256, 0, stream>>>(
      rz, Wd1, bd1, hdec, 2 * H_DIM + 1, 2 * H_DIM + 1, 0);
  k_decoder<<<(OUT_DIM + 255) / 256, 256, 0, stream>>>(
      hdec, Wd2, bd2, out, 2 * H_DIM + 1, OUT_DIM, 1);
}
